// ListWiseReformer_21552145891821
// MI455X (gfx1250) — matrix operand order repaired and checked by reading, not compiled, not run on hardware
//
#include <hip/hip_runtime.h>
#include <hip/hip_bf16.h>
#include <cstdint>
#include <cstddef>

// ---------------------------------------------------------------------------
// CDNA5 (gfx1250) Reformer forward. All heavy GEMMs run through
// v_wmma_f32_16x16x32_bf16 (wave32); GEMM tiles are staged into LDS with
// GLOBAL_LOAD_ASYNC_TO_LDS_B128 (ASYNCcnt) instead of VGPR round-trips.
// Weights pre-transposed to N x K so every WMMA fragment is a contiguous
// 16B LDS load (ds_load_b128).
// ---------------------------------------------------------------------------

typedef __attribute__((ext_vector_type(16))) __bf16 v16bf;
typedef __attribute__((ext_vector_type(8)))  __bf16 v8bf;
typedef __attribute__((ext_vector_type(8)))  float  v8f;

constexpr int B_   = 4;
constexpr int T_   = 4096;
constexpr int D_   = 512;
constexpr int H_   = 8;
constexpr int DH_  = 64;
constexpr int BH_  = B_ * H_;     // 32
constexpr int NH_  = 4;           // hash rounds
constexpr int NB_  = 64;          // buckets (rot_size)
constexpr int NCH_ = NH_ * NB_;   // 256 chunks per bh row
constexpr int NHT_ = NH_ * T_;    // 16384
constexpr int BT_  = B_ * T_;     // 16384 token rows
constexpr int FF_  = 4 * D_;      // 2048
constexpr int NPRED_ = 50;

#define WMMA_BF16(a, b, c) \
  __builtin_amdgcn_wmma_f32_16x16x32_bf16(false, (a), false, (b), (short)0, (c), false, false)

// Build a 16-element bf16 fragment from two aligned 16B chunks.
__device__ __forceinline__ v16bf ld_frag(const __bf16* p0, const __bf16* p1) {
  v8bf lo = *(const v8bf*)p0;
  v8bf hi = *(const v8bf*)p1;
  return __builtin_shufflevector(lo, hi, 0, 1, 2, 3, 4, 5, 6, 7,
                                 8, 9, 10, 11, 12, 13, 14, 15);
}

// Async copy of 32 contiguous bytes global -> LDS (two b128 transfers).
// IOFFSET applies to both the LDS and global address (ISA 15.18.3 / ch.10.7).
__device__ __forceinline__ void async_copy32(unsigned lds_off, const void* gaddr) {
  unsigned long long ga = (unsigned long long)gaddr;
  asm volatile("global_load_async_to_lds_b128 %0, %1, off"
               :: "v"(lds_off), "v"(ga) : "memory");
  asm volatile("global_load_async_to_lds_b128 %0, %1, off offset:16"
               :: "v"(lds_off), "v"(ga) : "memory");
}
__device__ __forceinline__ void wait_async0() {
  asm volatile("s_wait_asynccnt 0x0" ::: "memory");
}

// ---------------------------------------------------------------------------
// Weight f32 (K,N) -> bf16 transposed (N,K)
// ---------------------------------------------------------------------------
__global__ void transpose_bf16_kernel(const float* __restrict__ in,
                                      __bf16* __restrict__ out, int K, int N) {
  long total = (long)K * N;
  long i = (long)blockIdx.x * blockDim.x + threadIdx.x;
  long stride = (long)gridDim.x * blockDim.x;
  for (; i < total; i += stride) {
    int k = (int)(i / N), n = (int)(i % N);
    out[(size_t)n * K + k] = (__bf16)in[i];
  }
}

// ---------------------------------------------------------------------------
// Embedding: h = tok_emb[x] + pos_emb ; x1 = x2 = h
// ---------------------------------------------------------------------------
__global__ void embed_kernel(const int* __restrict__ x,
                             const float* __restrict__ tok,
                             const float* __restrict__ pos,
                             float* __restrict__ x1, float* __restrict__ x2) {
  int bt = blockIdx.x;
  int t  = bt & (T_ - 1);
  int token = x[bt];
  const float* tr = tok + (size_t)token * D_;
  const float* pr = pos + (size_t)t * D_;
  float* o1 = x1 + (size_t)bt * D_;
  float* o2 = x2 + (size_t)bt * D_;
  for (int d = threadIdx.x; d < D_; d += blockDim.x) {
    float h = tr[d] + pr[d];
    o1[d] = h; o2[d] = h;
  }
}

// ---------------------------------------------------------------------------
// LayerNorm: one wave32 per row of 512, bf16 output (GEMM input)
// ---------------------------------------------------------------------------
__global__ __launch_bounds__(256) void layernorm_kernel(
    const float* __restrict__ x, const float* __restrict__ g,
    const float* __restrict__ b, __bf16* __restrict__ out) {
  int row  = blockIdx.x * 8 + (threadIdx.x >> 5);
  int lane = threadIdx.x & 31;
  const float* xr = x + (size_t)row * D_;
  float v[16];
  float s = 0.f;
#pragma unroll
  for (int j = 0; j < 16; ++j) { v[j] = xr[lane + j * 32]; s += v[j]; }
#pragma unroll
  for (int o = 16; o > 0; o >>= 1) s += __shfl_xor(s, o, 32);
  float mu = s * (1.0f / D_);
  float vs = 0.f;
#pragma unroll
  for (int j = 0; j < 16; ++j) { float d = v[j] - mu; vs += d * d; }
#pragma unroll
  for (int o = 16; o > 0; o >>= 1) vs += __shfl_xor(vs, o, 32);
  float rstd = rsqrtf(vs * (1.0f / D_) + 1e-5f);
  __bf16* orow = out + (size_t)row * D_;
#pragma unroll
  for (int j = 0; j < 16; ++j) {
    int d = lane + j * 32;
    orow[d] = (__bf16)((v[j] - mu) * rstd * g[d] + b[d]);
  }
}

// ---------------------------------------------------------------------------
// Tiled bf16 WMMA GEMM: C[M,N] = act(A[M,K] @ W[K,N] + bias) (+res)
// A: (M,K) bf16 row-major. WT: (N,K) bf16 row-major (pre-transposed weight).
// Block tile 128x128, 8 waves (4 row-groups x 2 col-groups), each wave 32x64:
// 8 accumulators, 8 v_wmma per 32-wide K step. Tiles staged into dynamic LDS
// via global_load_async_to_lds_b128; fragments read as ds_load_b128.
// act: 0 = none, 1 = exact GELU. outF (f32) and/or outB (bf16) optional.
// M, N multiples of 128; K multiple of 32.
// ---------------------------------------------------------------------------
__global__ __launch_bounds__(256) void gemm_bf16_wmma(
    const __bf16* __restrict__ A, const __bf16* __restrict__ WT,
    const float* __restrict__ bias, const float* __restrict__ res,
    float* __restrict__ outF, __bf16* __restrict__ outB,
    int M, int N, int K, int act) {
  extern __shared__ char gsm[];
  __bf16* As = (__bf16*)gsm;          // [128][32] at LDS offset 0
  __bf16* Bs = As + 128 * 32;         // [128][32] at LDS offset 8192
  const int tid  = threadIdx.x;
  const int bm   = blockIdx.y * 128, bn = blockIdx.x * 128;
  const int wave = tid >> 5, lane = tid & 31;
  const int mi = wave & 3, ni = wave >> 2;   // 4 x 2 wave grid
  const int lr = lane & 15, lh = lane >> 4;
  const int ldr = tid >> 1, ldc = (tid & 1) << 4;   // tile-load coords
  const unsigned lds_a = (unsigned)(ldr * 64 + ldc * 2);          // bytes
  const unsigned lds_b = (unsigned)(8192 + ldr * 64 + ldc * 2);   // bytes

  v8f z = {};
  v8f acc[2][4] = {{z, z, z, z}, {z, z, z, z}};

  for (int k0 = 0; k0 < K; k0 += 32) {
    __syncthreads();
    // Async-stage A and B tiles (128x32 bf16 each): 2 x 32B per thread.
    const __bf16* ga = A + (size_t)(bm + ldr) * K + k0 + ldc;
    const __bf16* gb = WT + (size_t)(bn + ldr) * K + k0 + ldc;
    async_copy32(lds_a, ga);
    async_copy32(lds_b, gb);
    if (k0 + 32 < K) {   // prefetch next K slab (global_prefetch_b8)
      __builtin_prefetch(ga + 32, 0, 3);
      __builtin_prefetch(gb + 32, 0, 3);
    }
    wait_async0();
    __syncthreads();

    v16bf afr[2];
#pragma unroll
    for (int i = 0; i < 2; ++i) {
      int row = mi * 32 + i * 16 + lr;
      afr[i] = ld_frag(&As[row * 32 + lh * 8], &As[row * 32 + 16 + lh * 8]);
    }
#pragma unroll
    for (int s = 0; s < 4; ++s) {
      int col = ni * 64 + s * 16 + lr;
      v16bf bfr = ld_frag(&Bs[col * 32 + lh * 8], &Bs[col * 32 + 16 + lh * 8]);
#pragma unroll
      for (int i = 0; i < 2; ++i) acc[i][s] = WMMA_BF16(afr[i], bfr, acc[i][s]);
    }
  }

#pragma unroll
  for (int i = 0; i < 2; ++i) {
#pragma unroll
    for (int s = 0; s < 4; ++s) {
#pragma unroll
      for (int e = 0; e < 8; ++e) {
        int grow = bm + mi * 32 + i * 16 + lh * 8 + e;
        int gcol = bn + ni * 64 + s * 16 + lr;
        float v = acc[i][s][e];
        if (bias) v += bias[gcol];
        if (act == 1) v = 0.5f * v * (1.0f + erff(v * 0.70710678118f));
        if (res) v += res[(size_t)grow * N + gcol];
        if (outF) outF[(size_t)grow * N + gcol] = v;
        if (outB) outB[(size_t)grow * N + gcol] = (__bf16)v;
      }
    }
  }
}

// ---------------------------------------------------------------------------
// LSH bucketing: one wave per (bh,t,h). rotated = qk . rot, concat(-),
// argmax (first-max tie break), key = T*(argmax + h*NB) + t.
// ---------------------------------------------------------------------------
__global__ __launch_bounds__(256) void lsh_bucket_kernel(
    const float* __restrict__ qk, const float* __restrict__ rot,
    int* __restrict__ keys) {
  int gid  = blockIdx.x * 8 + (threadIdx.x >> 5);  // ((bh*T)+t)*NH + h
  int lane = threadIdx.x & 31;
  int h  = gid & 3;
  int t  = (gid >> 2) & (T_ - 1);
  int bh = gid >> 14;
  int b = bh >> 3, hd = bh & 7;
  const float* q  = qk + ((size_t)(b * T_ + t)) * D_ + hd * DH_;
  const float* rp = rot + h * 32 + lane;  // rot[f][h][r]: f-stride = NH*32
  float dot = 0.f;
#pragma unroll
  for (int f = 0; f < 64; ++f) dot += q[f] * rp[f * 128];
  float best = dot; int bi = lane;
  if (-dot > best) { best = -dot; bi = lane + 32; }
#pragma unroll
  for (int o = 16; o > 0; o >>= 1) {
    float ov = __shfl_xor(best, o, 32);
    int   oi = __shfl_xor(bi, o, 32);
    if (ov > best || (ov == best && oi < bi)) { best = ov; bi = oi; }
  }
  if (lane == 0) {
    int bucket = bi + h * NB_;
    keys[(size_t)bh * NHT_ + h * T_ + t] = T_ * bucket + t;
  }
}

// ---------------------------------------------------------------------------
// Bitonic argsort of 16384 unique keys per bh row in 128KB LDS (LDS=320KB).
// One workgroup (1024 threads) per row; emits sticker = argsort(keys).
// ---------------------------------------------------------------------------
__global__ __launch_bounds__(1024) void bitonic_sort_kernel(
    const int* __restrict__ keys, int* __restrict__ sticker) {
  extern __shared__ int sms[];
  int* k = sms;
  int* v = sms + NHT_;
  const int n = NHT_;
  int bh = blockIdx.x;
  for (int i = threadIdx.x; i < n; i += blockDim.x) {
    k[i] = keys[(size_t)bh * n + i];
    v[i] = i;
  }
  __syncthreads();
  for (int ks = 2; ks <= n; ks <<= 1) {
    for (int j = ks >> 1; j > 0; j >>= 1) {
      for (int i = threadIdx.x; i < n; i += blockDim.x) {
        int l = i ^ j;
        if (l > i) {
          bool up = ((i & ks) == 0);
          int ki = k[i], kl = k[l];
          bool sw = up ? (ki > kl) : (ki < kl);
          if (sw) {
            k[i] = kl; k[l] = ki;
            int tv = v[i]; v[i] = v[l]; v[l] = tv;
          }
        }
      }
      __syncthreads();
    }
  }
  for (int i = threadIdx.x; i < n; i += blockDim.x)
    sticker[(size_t)bh * n + i] = v[i];
}

// ---------------------------------------------------------------------------
// Chunked LSH attention, one block (8 waves) per (chunk, bh).
// bq(64x64), bk(128x64) row-major (K-contiguous), bv stored TRANSPOSED as
// bvT(64 dh x 128 key) so the P@V B-fragments are contiguous b128 loads.
// WMMA dots -> masked softmax/LSE -> WMMA probs@V; scatter rows by full
// sticker value (o[sticker[i]] = so[i] makes undo_sort implicit).
// ---------------------------------------------------------------------------
__global__ __launch_bounds__(256) void lsh_attn_kernel(
    const float* __restrict__ qk, const float* __restrict__ vv,
    const int* __restrict__ sticker, __bf16* __restrict__ o_bf,
    float* __restrict__ lse_out) {
  extern __shared__ char smem[];
  __bf16* bq   = (__bf16*)smem;              // 64*64
  __bf16* bk   = bq + 64 * 64;               // 128*64 (normalized keys)
  __bf16* bvT  = bk + 128 * 64;              // 64*128 (dh-major!)
  float*  dots = (float*)(bvT + 64 * 128);   // 64*128
  __bf16* pr   = (__bf16*)(dots + 64 * 128); // 64*128 probs
  int*  posk  = (int*)(pr + 64 * 128);       // 128 (st % T, cur then prev)
  int*  jfull = posk + 128;                  // 64 (full sticker of queries)

  const int c = blockIdx.x, bh = blockIdx.y;
  const int b = bh >> 3, h = bh & 7;
  const int tid  = threadIdx.x;
  const int wave = tid >> 5, lane = tid & 31;
  const int lr = lane & 15, lh = lane >> 4;

  if (tid < 64) {
    int sv = sticker[(size_t)bh * NHT_ + c * 64 + tid];
    posk[tid]  = sv & (T_ - 1);
    jfull[tid] = sv;
  } else if (tid < 128) {
    int pc = (c + NCH_ - 1) & (NCH_ - 1);     // circular look-one-back
    int sv = sticker[(size_t)bh * NHT_ + pc * 64 + (tid - 64)];
    posk[tid] = sv & (T_ - 1);
  }
  __syncthreads();

  // Gather qk/v rows; normalize keys (clip ||k|| >= 1e-12).
  for (int r = wave; r < 128; r += 8) {
    int pos = posk[r];
    const float* qr = qk + ((size_t)(b * T_ + pos)) * D_ + h * DH_;
    float x0 = qr[lane], x1 = qr[lane + 32];
    float ss = x0 * x0 + x1 * x1;
#pragma unroll
    for (int o = 16; o > 0; o >>= 1) ss += __shfl_xor(ss, o, 32);
    float inv = 1.0f / fmaxf(sqrtf(ss), 1e-12f);
    bk[r * 64 + lane]      = (__bf16)(x0 * inv);
    bk[r * 64 + lane + 32] = (__bf16)(x1 * inv);
    if (r < 64) {
      bq[r * 64 + lane]      = (__bf16)x0;
      bq[r * 64 + lane + 32] = (__bf16)x1;
    }
    const float* vr = vv + ((size_t)(b * T_ + pos)) * D_ + h * DH_;
    bvT[lane * 128 + r]        = (__bf16)vr[lane];        // transposed store
    bvT[(lane + 32) * 128 + r] = (__bf16)vr[lane + 32];
  }
  __syncthreads();

  // dots = bq @ bk^T * DH^-0.5, self-masked to -5e4. 32 tiles / 8 waves.
  {
    const int mi = wave & 3;
    const int ng = (wave >> 2) * 4;
    v8f z = {};
    v8f acc[4] = {z, z, z, z};
#pragma unroll
    for (int kb = 0; kb < 64; kb += 32) {
      int row = mi * 16 + lr;
      v16bf a = ld_frag(&bq[row * 64 + kb + lh * 8],
                        &bq[row * 64 + kb + 16 + lh * 8]);
#pragma unroll
      for (int s = 0; s < 4; ++s) {
        int col = (ng + s) * 16 + lr;      // key index (B column)
        v16bf bb = ld_frag(&bk[col * 64 + kb + lh * 8],
                           &bk[col * 64 + kb + 16 + lh * 8]);
        acc[s] = WMMA_BF16(a, bb, acc[s]);
      }
    }
#pragma unroll
    for (int s = 0; s < 4; ++s) {
#pragma unroll
      for (int e = 0; e < 8; ++e) {
        int row = mi * 16 + lh * 8 + e;
        int col = (ng + s) * 16 + lr;
        float dv = acc[s][e] * 0.125f;     // DH^-0.5
        if (posk[row] == posk[col]) dv = -5e4f;
        dots[row * 128 + col] = dv;
      }
    }
  }
  __syncthreads();

  // Row softmax (128 wide) + LSE; probs -> bf16. LSE scattered by jfull.
  for (int r = wave; r < 64; r += 8) {
    float vals[4];
#pragma unroll
    for (int j = 0; j < 4; ++j) vals[j] = dots[r * 128 + lane + j * 32];
    float m = fmaxf(fmaxf(vals[0], vals[1]), fmaxf(vals[2], vals[3]));
#pragma unroll
    for (int o = 16; o > 0; o >>= 1) m = fmaxf(m, __shfl_xor(m, o, 32));
    float s = 0.f;
#pragma unroll
    for (int j = 0; j < 4; ++j) s += expf(vals[j] - m);
#pragma unroll
    for (int o = 16; o > 0; o >>= 1) s += __shfl_xor(s, o, 32);
    float lsev = logf(s) + m;
#pragma unroll
    for (int j = 0; j < 4; ++j)
      pr[r * 128 + lane + j * 32] = (__bf16)expf(vals[j] - lsev);
    if (lane == 0) lse_out[(size_t)bh * NHT_ + jfull[r]] = lsev;
  }
  __syncthreads();

  // bo = probs(64x128) @ bv(128x64). 16 tiles / 8 waves. Scatter by jfull.
  {
    const int mi = wave & 3;
    const int ng = (wave >> 2) * 2;
    v8f z = {};
    v8f acc[2] = {z, z};
#pragma unroll
    for (int kb = 0; kb < 128; kb += 32) {
      int row = mi * 16 + lr;
      v16bf a = ld_frag(&pr[row * 128 + kb + lh * 8],
                        &pr[row * 128 + kb + 16 + lh * 8]);
#pragma unroll
      for (int s = 0; s < 2; ++s) {
        int col = (ng + s) * 16 + lr;      // dh index (B column)
        v16bf bb = ld_frag(&bvT[col * 128 + kb + lh * 8],
                           &bvT[col * 128 + kb + 16 + lh * 8]);
        acc[s] = WMMA_BF16(a, bb, acc[s]);
      }
    }
#pragma unroll
    for (int s = 0; s < 2; ++s) {
#pragma unroll
      for (int e = 0; e < 8; ++e) {
        int row = mi * 16 + lh * 8 + e;
        int dh  = (ng + s) * 16 + lr;
        int jj  = jfull[row];
        o_bf[((size_t)bh * NHT_ + jj) * DH_ + dh] = (__bf16)acc[s][e];
      }
    }
  }
}

// ---------------------------------------------------------------------------
// Combine NH hash rounds with softmax(lse) weights; write (B,T,D) bf16.
// ---------------------------------------------------------------------------
__global__ __launch_bounds__(256) void combine_kernel(
    const __bf16* __restrict__ o_bf, const float* __restrict__ lse,
    __bf16* __restrict__ attn_bf) {
  int idx = blockIdx.x * 4 + (threadIdx.x >> 6);   // bh*T + t
  int d   = threadIdx.x & 63;
  int t   = idx & (T_ - 1);
  int bh  = idx >> 12;
  float l[NH_];
#pragma unroll
  for (int hh = 0; hh < NH_; ++hh) l[hh] = lse[(size_t)bh * NHT_ + hh * T_ + t];
  float mx = fmaxf(fmaxf(l[0], l[1]), fmaxf(l[2], l[3]));
  float w[NH_], sum = 0.f;
#pragma unroll
  for (int hh = 0; hh < NH_; ++hh) { w[hh] = expf(l[hh] - mx); sum += w[hh]; }
  float inv = 1.0f / sum;
  float acc = 0.f;
#pragma unroll
  for (int hh = 0; hh < NH_; ++hh)
    acc += w[hh] * inv *
           (float)o_bf[((size_t)bh * NHT_ + hh * T_ + t) * DH_ + d];
  int b = bh >> 3, hd = bh & 7;
  attn_bf[((size_t)(b * T_ + t)) * D_ + hd * DH_ + d] = (__bf16)acc;
}

// ---------------------------------------------------------------------------
// Final head (tiny): pooled = tanh(out[:,0] @ pool_w + pool_b); logits.
// ---------------------------------------------------------------------------
__global__ void pool_kernel(const float* __restrict__ x1,
                            const float* __restrict__ x2,
                            const float* __restrict__ pw,
                            const float* __restrict__ pb,
                            float* __restrict__ pooled) {
  int b = blockIdx.x;
  const float* r1 = x1 + (size_t)b * T_ * D_;   // t = 0 row
  const float* r2 = x2 + (size_t)b * T_ * D_;
  for (int n = threadIdx.x; n < D_; n += blockDim.x) {
    float acc = pb[n];
    for (int d = 0; d < D_; ++d)
      acc += 0.5f * (r1[d] + r2[d]) * pw[(size_t)d * D_ + n];
    pooled[b * D_ + n] = tanhf(acc);
  }
}

__global__ void logits_kernel(const float* __restrict__ pooled,
                              const float* __restrict__ lw,
                              const float* __restrict__ lb,
                              float* __restrict__ out) {
  int b = blockIdx.x;
  int n = threadIdx.x;
  if (n < NPRED_) {
    float acc = lb[n];
    for (int d = 0; d < D_; ++d)
      acc += pooled[b * D_ + d] * lw[(size_t)d * NPRED_ + n];
    out[b * NPRED_ + n] = acc;
  }
}

// ---------------------------------------------------------------------------
// Orchestration
// ---------------------------------------------------------------------------
extern "C" void kernel_launch(void* const* d_in, const int* in_sizes, int n_in,
                              void* d_out, int out_size, void* d_ws,
                              size_t ws_size, hipStream_t stream) {
  const int*   x        = (const int*)d_in[0];
  const float* tok_emb  = (const float*)d_in[1];
  const float* pos_emb  = (const float*)d_in[2];
  const float* wqk      = (const float*)d_in[3];
  const float* wv       = (const float*)d_in[4];
  const float* wout_w   = (const float*)d_in[5];
  const float* wout_b   = (const float*)d_in[6];
  const float* ln1_g    = (const float*)d_in[7];
  const float* ln1_b    = (const float*)d_in[8];
  const float* ln2_g    = (const float*)d_in[9];
  const float* ln2_b    = (const float*)d_in[10];
  const float* ff_w1    = (const float*)d_in[11];
  const float* ff_b1    = (const float*)d_in[12];
  const float* ff_w2    = (const float*)d_in[13];
  const float* ff_b2    = (const float*)d_in[14];
  const float* pool_w   = (const float*)d_in[15];
  const float* pool_b   = (const float*)d_in[16];
  const float* logit_w  = (const float*)d_in[17];
  const float* logit_b  = (const float*)d_in[18];
  const float* rotations = (const float*)d_in[19];
  float* out = (float*)d_out;

  char* ws = (char*)d_ws;
  size_t off = 0;
  auto carve = [&](size_t bytes) -> void* {
    void* p = ws + off;
    off += (bytes + 255) & ~(size_t)255;
    return p;
  };

  float*  x1f    = (float*) carve((size_t)BT_ * D_ * 4);
  float*  x2f    = (float*) carve((size_t)BT_ * D_ * 4);
  __bf16* lnbf   = (__bf16*)carve((size_t)BT_ * D_ * 2);
  float*  qkf    = (float*) carve((size_t)BT_ * D_ * 4);
  float*  vvf    = (float*) carve((size_t)BT_ * D_ * 4);
  int*    keys   = (int*)   carve((size_t)BH_ * NHT_ * 4);
  int*    stick  = (int*)   carve((size_t)BH_ * NHT_ * 4);
  __bf16* obf    = (__bf16*)carve((size_t)BH_ * NHT_ * DH_ * 2);
  float*  lsef   = (float*) carve((size_t)BH_ * NHT_ * 4);
  __bf16* attnbf = (__bf16*)carve((size_t)BT_ * D_ * 2);
  __bf16* ffbf   = (__bf16*)carve((size_t)BT_ * FF_ * 2);
  __bf16* wqkT   = (__bf16*)carve((size_t)2 * D_ * D_ * 2);   // (N,K) bf16
  __bf16* wvT    = (__bf16*)carve((size_t)2 * D_ * D_ * 2);
  __bf16* woutT  = (__bf16*)carve((size_t)2 * D_ * D_ * 2);
  __bf16* w1T    = (__bf16*)carve((size_t)2 * D_ * FF_ * 2);
  __bf16* w2T    = (__bf16*)carve((size_t)2 * FF_ * D_ * 2);
  float*  pooled = (float*) carve((size_t)B_ * D_ * 4);

  // Weights -> bf16, transposed to (N,K), per layer (deterministic).
  for (int l = 0; l < 2; ++l) {
    transpose_bf16_kernel<<<512, 256, 0, stream>>>(
        wqk + (size_t)l * D_ * D_, wqkT + (size_t)l * D_ * D_, D_, D_);
    transpose_bf16_kernel<<<512, 256, 0, stream>>>(
        wv + (size_t)l * D_ * D_, wvT + (size_t)l * D_ * D_, D_, D_);
    transpose_bf16_kernel<<<512, 256, 0, stream>>>(
        wout_w + (size_t)l * D_ * D_, woutT + (size_t)l * D_ * D_, D_, D_);
    transpose_bf16_kernel<<<1024, 256, 0, stream>>>(
        ff_w1 + (size_t)l * D_ * FF_, w1T + (size_t)l * D_ * FF_, D_, FF_);
    transpose_bf16_kernel<<<1024, 256, 0, stream>>>(
        ff_w2 + (size_t)l * FF_ * D_, w2T + (size_t)l * FF_ * D_, FF_, D_);
  }

  embed_kernel<<<BT_, 256, 0, stream>>>(x, tok_emb, pos_emb, x1f, x2f);

  constexpr size_t GEMM_SMEM = (size_t)2 * 128 * 32 * 2;   // 16 KB
  constexpr size_t ATTN_SMEM =
      (size_t)64 * 64 * 2 + (size_t)128 * 64 * 2 + (size_t)64 * 128 * 2 +
      (size_t)64 * 128 * 4 + (size_t)64 * 128 * 2 + 128 * 4 + 64 * 4;
  constexpr size_t SORT_SMEM = (size_t)2 * NHT_ * 4;   // 128 KB (LDS=320KB)

  for (int l = 0; l < 2; ++l) {
    // --- attention block ---
    layernorm_kernel<<<BT_ / 8, 256, 0, stream>>>(x2f, ln1_g + l * D_,
                                                  ln1_b + l * D_, lnbf);
    gemm_bf16_wmma<<<dim3(D_ / 128, BT_ / 128), 256, GEMM_SMEM, stream>>>(
        lnbf, wqkT + (size_t)l * D_ * D_, nullptr, nullptr, qkf, nullptr,
        BT_, D_, D_, 0);
    gemm_bf16_wmma<<<dim3(D_ / 128, BT_ / 128), 256, GEMM_SMEM, stream>>>(
        lnbf, wvT + (size_t)l * D_ * D_, nullptr, nullptr, vvf, nullptr,
        BT_, D_, D_, 0);

    lsh_bucket_kernel<<<(BH_ * T_ * NH_) / 8, 256, 0, stream>>>(
        qkf, rotations + (size_t)l * DH_ * NH_ * (NB_ / 2), keys);
    bitonic_sort_kernel<<<BH_, 1024, SORT_SMEM, stream>>>(keys, stick);

    lsh_attn_kernel<<<dim3(NCH_, BH_), 256, ATTN_SMEM, stream>>>(
        qkf, vvf, stick, obf, lsef);
    combine_kernel<<<(BH_ * T_) / 4, 256, 0, stream>>>(obf, lsef, attnbf);

    // y1 = x1 + attn_out @ wout + b   (in-place into x1)
    gemm_bf16_wmma<<<dim3(D_ / 128, BT_ / 128), 256, GEMM_SMEM, stream>>>(
        attnbf, woutT + (size_t)l * D_ * D_, wout_b + l * D_, x1f, x1f,
        nullptr, BT_, D_, D_, 0);

    // --- feed-forward block ---
    layernorm_kernel<<<BT_ / 8, 256, 0, stream>>>(x1f, ln2_g + l * D_,
                                                  ln2_b + l * D_, lnbf);
    gemm_bf16_wmma<<<dim3(FF_ / 128, BT_ / 128), 256, GEMM_SMEM, stream>>>(
        lnbf, w1T + (size_t)l * D_ * FF_, ff_b1 + l * FF_, nullptr, nullptr,
        ffbf, BT_, FF_, D_, 1);  // GELU epilogue, bf16 out
    // y2 = x2 + gelu(...) @ w2 + b2   (in-place into x2)
    gemm_bf16_wmma<<<dim3(D_ / 128, BT_ / 128), 256, GEMM_SMEM, stream>>>(
        ffbf, w2T + (size_t)l * FF_ * D_, ff_b2 + l * D_, x2f, x2f, nullptr,
        BT_, D_, FF_, 0);
  }

  pool_kernel<<<B_, 256, 0, stream>>>(x1f, x2f, pool_w, pool_b, pooled);
  logits_kernel<<<B_, 64, 0, stream>>>(pooled, logit_w, logit_b, out);
}
